// Attention_10213432229985
// MI455X (gfx1250) — matrix operand order repaired and checked by reading, not compiled, not run on hardware
//
#include <hip/hip_runtime.h>
#include <hip/hip_bf16.h>

// ---------------------------------------------------------------------------
// Problem constants (reference: B=8, C=512, T=4096, fp32 in/out)
// ---------------------------------------------------------------------------
#define BATCH 8
#define CH    512
#define TT    4096
#define SCALE 0.125f  // 64^-0.5

typedef __attribute__((ext_vector_type(16))) __bf16 v16bf;
typedef __attribute__((ext_vector_type(8)))  __bf16 v8bf;
typedef __attribute__((ext_vector_type(8)))  float  v8f;

// ---------------------------------------------------------------------------
// WMMA helper (CDNA5 gfx1250, wave32): D = A(16x32 bf16) x B(32x16 bf16) + C
// ---------------------------------------------------------------------------
__device__ __forceinline__ v8f wmma_bf16(v16bf a, v16bf b, v8f c) {
    return __builtin_amdgcn_wmma_f32_16x16x32_bf16(
        /*neg_a=*/false, a, /*neg_b=*/false, b,
        /*c_mod=*/(short)0, c, /*reuse_a=*/false, /*reuse_b=*/false);
}

// A fragment: tile row-major in LDS, 16 rows (M) x 32 (K), ld elems/row.
// Lane half h, row = lane&15, K in {8h..8h+7} u {16+8h..16+8h+7}
// -> two contiguous 16B chunks -> 2x ds_load_b128.
__device__ __forceinline__ v16bf load_a_frag(const __bf16* s, int ld) {
    const int lane = threadIdx.x & 31;
    const int row  = lane & 15;
    const int h    = lane >> 4;
    const __bf16* rp = s + row * ld;
    const v8bf lo = *(const v8bf*)(rp + 8 * h);
    const v8bf hi = *(const v8bf*)(rp + 16 + 8 * h);
    v16bf a;
#pragma unroll
    for (int j = 0; j < 8; ++j) { a[j] = lo[j]; a[8 + j] = hi[j]; }
    return a;
}

// B fragment from an n-major tile: element [k][n] stored at s[n*ld + k].
// Lane half h holds K in {8h..8h+7} u {16+8h..16+8h+7} for column n = lane&15
// -> two contiguous 16B chunks -> 2x ds_load_b128.
__device__ __forceinline__ v16bf load_b_fragT(const __bf16* s, int ld) {
    const int lane = threadIdx.x & 31;
    const int n    = lane & 15;
    const int h    = lane >> 4;
    const __bf16* rp = s + n * ld;
    const v8bf lo = *(const v8bf*)(rp + 8 * h);
    const v8bf hi = *(const v8bf*)(rp + 16 + 8 * h);
    v16bf bv;
#pragma unroll
    for (int j = 0; j < 8; ++j) { bv[j] = lo[j]; bv[8 + j] = hi[j]; }
    return bv;
}

// ---------------------------------------------------------------------------
// Async global->LDS 16B copy (GLOBAL_LOAD_ASYNC_TO_LDS_B128, ASYNCcnt).
// LDS address = low 32 bits of the flat shared address (HW truncates).
// ---------------------------------------------------------------------------
__device__ __forceinline__ void async_ld_b128(void* lds, const void* gptr) {
    const unsigned l = (unsigned)(size_t)lds;
    asm volatile("global_load_async_to_lds_b128 %0, %1, off"
                 :: "v"(l), "v"(gptr) : "memory");
}
template <int N>
__device__ __forceinline__ void wait_async_le() {
#if defined(__has_builtin) && __has_builtin(__builtin_amdgcn_s_wait_asynccnt)
    __builtin_amdgcn_s_wait_asynccnt(N);
#else
    asm volatile("s_wait_asynccnt %0" :: "n"(N) : "memory");
#endif
}

// Pack two f32 -> adjacent bf16 pair, single b32 store (v_cvt_pk_bf16_f32).
__device__ __forceinline__ void store_pack_bf16(__bf16* p, float a, float b) {
    union { unsigned u; __bf16 h[2]; } t;
    t.h[0] = (__bf16)a;
    t.h[1] = (__bf16)b;
    *(unsigned*)p = t.u;
}

// ---------------------------------------------------------------------------
// Kernel 1: fused QKV projection. q/k/v = W(512x512) x x(512x4096) + b
// block = 256 (8 waves). Block tile 64(O) x 64(T); wave tile 16(O) x 32(T).
// x tile staged n-major [t=64][k=32]; W tiles row-major [o=64][k=32].
// ---------------------------------------------------------------------------
__global__ __launch_bounds__(256)
void qkv_kernel(const float* __restrict__ x,
                const float* __restrict__ Wq, const float* __restrict__ bq,
                const float* __restrict__ Wk, const float* __restrict__ bk,
                const float* __restrict__ Wv, const float* __restrict__ bv,
                __bf16* __restrict__ qo, __bf16* __restrict__ ko,
                __bf16* __restrict__ vo) {
    const int b  = blockIdx.z;
    const int o0 = blockIdx.y * 64;
    const int t0 = blockIdx.x * 64;
    const int tid  = threadIdx.x;
    const int wave = tid >> 5;
    const int wo   = wave & 3;   // O strip of 16
    const int wt   = wave >> 2;  // T strip of 32

    __shared__ __align__(32) __bf16 xs[64 * 32];   // [t][k]  (n-major)
    __shared__ __align__(32) __bf16 wqs[64 * 32];  // [o][k]  (row-major)
    __shared__ __align__(32) __bf16 wks[64 * 32];
    __shared__ __align__(32) __bf16 wvs[64 * 32];

    v8f aq0 = {}, aq1 = {}, ak0 = {}, ak1 = {}, av0 = {}, av1 = {};

    for (int cc = 0; cc < CH; cc += 32) {
        // stage x chunk: two coalesced f32 b128 loads per thread (rows k,k+1),
        // packed (k,k+1) bf16-pair b32 stores into the n-major tile.
        {
            const int tc4 = (tid & 15) * 4;   // t offset 0..60
            const int kp  = (tid >> 4) * 2;   // k pair base 0..30
            const size_t rbase = ((size_t)b * CH + cc + kp) * TT + t0 + tc4;
            const float4 f0 = *(const float4*)&x[rbase];
            const float4 f1 = *(const float4*)&x[rbase + TT];
            store_pack_bf16(&xs[(tc4 + 0) * 32 + kp], f0.x, f1.x);
            store_pack_bf16(&xs[(tc4 + 1) * 32 + kp], f0.y, f1.y);
            store_pack_bf16(&xs[(tc4 + 2) * 32 + kp], f0.z, f1.z);
            store_pack_bf16(&xs[(tc4 + 3) * 32 + kp], f0.w, f1.w);
            if (cc + 32 < CH)
                __builtin_prefetch(&x[rbase + (size_t)32 * TT], 0, 3);
        }
        // stage weight chunks: f32 b128 loads, row-major bf16 stores (b64 packs)
        {
            const int kk4 = (tid & 7) * 4;  // k offset 0..28
            const int r   = tid >> 3;       // o row 0..31 (x2 halves)
#pragma unroll
            for (int half = 0; half < 2; ++half) {
                const int o = r + 32 * half;
                const size_t off = (size_t)(o0 + o) * CH + cc + kk4;
                const float4 fq = *(const float4*)&Wq[off];
                const float4 fk = *(const float4*)&Wk[off];
                const float4 fv = *(const float4*)&Wv[off];
                store_pack_bf16(&wqs[o * 32 + kk4],     fq.x, fq.y);
                store_pack_bf16(&wqs[o * 32 + kk4 + 2], fq.z, fq.w);
                store_pack_bf16(&wks[o * 32 + kk4],     fk.x, fk.y);
                store_pack_bf16(&wks[o * 32 + kk4 + 2], fk.z, fk.w);
                store_pack_bf16(&wvs[o * 32 + kk4],     fv.x, fv.y);
                store_pack_bf16(&wvs[o * 32 + kk4 + 2], fv.z, fv.w);
            }
        }
        __syncthreads();

        const v16bf aWq = load_a_frag(wqs + wo * 16 * 32, 32);
        const v16bf aWk = load_a_frag(wks + wo * 16 * 32, 32);
        const v16bf aWv = load_a_frag(wvs + wo * 16 * 32, 32);
        const v16bf bx0 = load_b_fragT(xs + (wt * 32) * 32, 32);
        const v16bf bx1 = load_b_fragT(xs + (wt * 32 + 16) * 32, 32);

        aq0 = wmma_bf16(aWq, bx0, aq0);
        aq1 = wmma_bf16(aWq, bx1, aq1);
        ak0 = wmma_bf16(aWk, bx0, ak0);
        ak1 = wmma_bf16(aWk, bx1, ak1);
        av0 = wmma_bf16(aWv, bx0, av0);
        av1 = wmma_bf16(aWv, bx1, av1);
        __syncthreads();
    }

    // C/D layout: VGPR r -> row m = r + 8*half, col n = lane&15
    const int lane = tid & 31;
    const int n = lane & 15;
    const int h = lane >> 4;
#pragma unroll
    for (int r = 0; r < 8; ++r) {
        const int o = o0 + wo * 16 + r + 8 * h;
        const size_t base = ((size_t)b * CH + o) * TT + t0 + wt * 32 + n;
        const float biq = bq[o], bik = bk[o], biv = bv[o];
        qo[base]      = (__bf16)(aq0[r] + biq);
        qo[base + 16] = (__bf16)(aq1[r] + biq);
        ko[base]      = (__bf16)(ak0[r] + bik);
        ko[base + 16] = (__bf16)(ak1[r] + bik);
        vo[base]      = (__bf16)(av0[r] + biv);
        vo[base + 16] = (__bf16)(av1[r] + biv);
    }
}

// ---------------------------------------------------------------------------
// Kernel 2: dots[b,c,d] = SCALE * sum_t q[b,c,t] * k[b,d,t]   (K = 4096)
// Double-buffered async pipeline: GLOBAL_LOAD_ASYNC_TO_LDS_B128 for chunk i+1
// overlaps with WMMA on chunk i; s_wait_asynccnt 2 drains only chunk i
// (async completions are in-order per wave).
// ---------------------------------------------------------------------------
__global__ __launch_bounds__(256)
void dots_kernel(const __bf16* __restrict__ qb, const __bf16* __restrict__ kb,
                 float* __restrict__ dots) {
    const int b  = blockIdx.z;
    const int c0 = blockIdx.y * 64;
    const int d0 = blockIdx.x * 64;
    const int tid  = threadIdx.x;
    const int wave = tid >> 5;
    const int wc   = wave & 3;   // c strip of 16
    const int wd   = wave >> 2;  // d strip of 32

    __shared__ __align__(32) __bf16 qs[2][64 * 32];  // [c][t]
    __shared__ __align__(32) __bf16 ks[2][64 * 32];  // [d][t] == n-major for B

    v8f a0 = {}, a1 = {};

    const int r   = tid >> 2;        // row 0..63
    const int cc8 = (tid & 3) << 3;  // bf16 offset 0,8,16,24 (16B chunks)
    const __bf16* qrow = qb + ((size_t)b * CH + c0 + r) * TT + cc8;
    const __bf16* krow = kb + ((size_t)b * CH + d0 + r) * TT + cc8;
    const int lofs = r * 32 + cc8;

    // prologue: chunk 0 -> buffer 0
    async_ld_b128(&qs[0][lofs], qrow);
    async_ld_b128(&ks[0][lofs], krow);

    const int NIT = TT / 32;  // 128
    for (int i = 0; i < NIT; ++i) {
        const int cur = i & 1;
        if (i + 1 < NIT) {
            const int nxt = cur ^ 1;
            async_ld_b128(&qs[nxt][lofs], qrow + (size_t)(i + 1) * 32);
            async_ld_b128(&ks[nxt][lofs], krow + (size_t)(i + 1) * 32);
            wait_async_le<2>();  // chunk i landed; chunk i+1 still in flight
        } else {
            wait_async_le<0>();
        }
        __syncthreads();

        const v16bf af = load_a_frag(qs[cur] + wc * 16 * 32, 32);
        const v16bf b0 = load_b_fragT(ks[cur] + (wd * 32) * 32, 32);
        const v16bf b1 = load_b_fragT(ks[cur] + (wd * 32 + 16) * 32, 32);
        a0 = wmma_bf16(af, b0, a0);
        a1 = wmma_bf16(af, b1, a1);
        __syncthreads();
    }

    const int lane = tid & 31;
    const int n = lane & 15;
    const int h = lane >> 4;
#pragma unroll
    for (int rr = 0; rr < 8; ++rr) {
        const int c = c0 + wc * 16 + rr + 8 * h;
        const size_t base = ((size_t)b * CH + c) * CH + d0 + wd * 32 + n;
        dots[base]      = a0[rr] * SCALE;
        dots[base + 16] = a1[rr] * SCALE;
    }
}

// ---------------------------------------------------------------------------
// Kernel 3: row softmax over 512 entries; attn out as bf16.
// ---------------------------------------------------------------------------
__global__ __launch_bounds__(256)
void softmax_kernel(const float* __restrict__ dots, __bf16* __restrict__ attn) {
    const int row = blockIdx.x;  // b*512 + c
    const float* rp = dots + (size_t)row * CH;
    __bf16* ap = attn + (size_t)row * CH;
    const int tid = threadIdx.x;
    const int wave = tid >> 5, lane = tid & 31;

    __shared__ float red[8];
    __shared__ float smax, ssum;

    const float v0 = rp[tid];
    const float v1 = rp[tid + 256];

    float m = fmaxf(v0, v1);
#pragma unroll
    for (int off = 16; off; off >>= 1) m = fmaxf(m, __shfl_xor(m, off, 32));
    if (lane == 0) red[wave] = m;
    __syncthreads();
    if (tid == 0) {
        float mm = red[0];
#pragma unroll
        for (int i = 1; i < 8; ++i) mm = fmaxf(mm, red[i]);
        smax = mm;
    }
    __syncthreads();

    const float e0 = __expf(v0 - smax);
    const float e1 = __expf(v1 - smax);
    float s = e0 + e1;
#pragma unroll
    for (int off = 16; off; off >>= 1) s += __shfl_xor(s, off, 32);
    if (lane == 0) red[wave] = s;
    __syncthreads();
    if (tid == 0) {
        float ss = red[0];
#pragma unroll
        for (int i = 1; i < 8; ++i) ss += red[i];
        ssum = ss;
    }
    __syncthreads();

    const float inv = 1.0f / ssum;
    ap[tid]       = (__bf16)(e0 * inv);
    ap[tid + 256] = (__bf16)(e1 * inv);
}

// ---------------------------------------------------------------------------
// Kernel 4: out[b,c,t] = sum_d attn[b,c,d] * v[b,d,t]   (K = 512), f32 out.
// attn A-tile: async b128 raw copy (row-major fits A frags).
// v B-tile: one coalesced 16B global load per thread along t, then 8 b16 LDS
// stores transposing into the n-major [t=64][d=32] layout.
// ---------------------------------------------------------------------------
__global__ __launch_bounds__(256)
void av_kernel(const __bf16* __restrict__ attn, const __bf16* __restrict__ vb,
               float* __restrict__ out) {
    const int b  = blockIdx.z;
    const int c0 = blockIdx.y * 64;
    const int t0 = blockIdx.x * 64;
    const int tid  = threadIdx.x;
    const int wave = tid >> 5;
    const int wc   = wave & 3;   // c strip of 16
    const int wt   = wave >> 2;  // t strip of 32

    __shared__ __align__(32) __bf16 as_[64 * 32];  // [c][d]  row-major (A)
    __shared__ __align__(32) __bf16 vs[64 * 32];   // [t][d]  n-major  (B)

    v8f a0 = {}, a1 = {};

    const int ar  = tid >> 2;        // attn row 0..63
    const int ac8 = (tid & 3) << 3;  // 16B chunk offset
    const int tc8 = (tid & 7) * 8;   // t offset 0..56
    const int dr  = tid >> 3;        // d row 0..31

    for (int dd = 0; dd < CH; dd += 32) {
        // A tile: raw async copy, attn[c0+ar][dd..dd+32)
        async_ld_b128(as_ + ar * 32 + ac8,
                      attn + ((size_t)b * CH + c0 + ar) * CH + dd + ac8);
        // B tile: coalesced b128 load along t, transpose into vs[t][d]
        {
            const v8bf vv = *(const v8bf*)&vb[((size_t)b * CH + dd + dr) * TT + t0 + tc8];
#pragma unroll
            for (int j = 0; j < 8; ++j)
                vs[(tc8 + j) * 32 + dr] = vv[j];
        }
        wait_async_le<0>();
        __syncthreads();

        const v16bf af = load_a_frag(as_ + wc * 16 * 32, 32);
        const v16bf b0 = load_b_fragT(vs + (wt * 32) * 32, 32);
        const v16bf b1 = load_b_fragT(vs + (wt * 32 + 16) * 32, 32);
        a0 = wmma_bf16(af, b0, a0);
        a1 = wmma_bf16(af, b1, a1);
        __syncthreads();
    }

    const int lane = tid & 31;
    const int n = lane & 15;
    const int h = lane >> 4;
#pragma unroll
    for (int rr = 0; rr < 8; ++rr) {
        const int c = c0 + wc * 16 + rr + 8 * h;
        const size_t base = ((size_t)b * CH + c) * TT + t0 + wt * 32 + n;
        out[base]      = a0[rr];
        out[base + 16] = a1[rr];
    }
}

// ---------------------------------------------------------------------------
// Launch. Workspace layout (bytes):
//   [0,   32M)  q  bf16   B*C*T
//   [32M, 64M)  k  bf16
//   [64M, 96M)  v  bf16
//   [96M, 104M) dots f32  B*C*C
//   [104M,108M) attn bf16 B*C*C
// ---------------------------------------------------------------------------
extern "C" void kernel_launch(void* const* d_in, const int* in_sizes, int n_in,
                              void* d_out, int out_size, void* d_ws, size_t ws_size,
                              hipStream_t stream) {
    const float* x  = (const float*)d_in[0];
    const float* Wq = (const float*)d_in[1];
    const float* bq = (const float*)d_in[2];
    const float* Wk = (const float*)d_in[3];
    const float* bk = (const float*)d_in[4];
    const float* Wv = (const float*)d_in[5];
    const float* bv = (const float*)d_in[6];
    float* out = (float*)d_out;

    const size_t qkvBytes  = (size_t)BATCH * CH * TT * sizeof(__bf16);  // 32 MB
    const size_t dotsBytes = (size_t)BATCH * CH * CH * sizeof(float);   //  8 MB

    char* ws = (char*)d_ws;
    __bf16* qb   = (__bf16*)(ws);
    __bf16* kb   = (__bf16*)(ws + qkvBytes);
    __bf16* vb   = (__bf16*)(ws + 2 * qkvBytes);
    float*  dots = (float*)(ws + 3 * qkvBytes);
    __bf16* attn = (__bf16*)(ws + 3 * qkvBytes + dotsBytes);

    dim3 blk(256);
    dim3 gQKV(TT / 64, CH / 64, BATCH);  // (64, 8, 8)
    dim3 gDot(CH / 64, CH / 64, BATCH);  // (8, 8, 8)
    dim3 gSft(BATCH * CH);               // 4096
    dim3 gAV(TT / 64, CH / 64, BATCH);   // (64, 8, 8)

    qkv_kernel<<<gQKV, blk, 0, stream>>>(x, Wq, bq, Wk, bk, Wv, bv, qb, kb, vb);
    dots_kernel<<<gDot, blk, 0, stream>>>(qb, kb, dots);
    softmax_kernel<<<gSft, blk, 0, stream>>>(dots, attn);
    av_kernel<<<gAV, blk, 0, stream>>>(attn, vb, out);
}
